// MSDeformAttn_57166014710110
// MI455X (gfx1250) — matrix operand order repaired and checked by reading, compiled, not run on hardware
//
#include <hip/hip_runtime.h>
#include <hip/hip_bf16.h>

// ---------------------------------------------------------------------------
// MS-Deformable-Attention for MI455X (gfx1250, wave32, WMMA bf16 path)
// ---------------------------------------------------------------------------
// Shapes (from reference): Lq = Lin = 21760, D_MODEL = 256, heads=8, levels=4,
// points=4, Dh=32.  SPATIAL = {128x128, 64x64, 32x32, 16x16},
// level starts = {0, 16384, 20480, 21504}.

typedef __attribute__((ext_vector_type(16))) __bf16 v16bf;
typedef __attribute__((ext_vector_type(8)))  __bf16 v8bf;
typedef __attribute__((ext_vector_type(8)))  float  v8f;

#define LQ      21760
#define LIN     21760
#define DM      256
#define NHEADS  8
#define DH      32

// ---------------------------------------------------------------------------
// fp32 -> bf16 elementwise convert
// ---------------------------------------------------------------------------
__global__ __launch_bounds__(256) void f32_to_bf16_kernel(
    const float* __restrict__ in, __bf16* __restrict__ out, int n)
{
    int i = blockIdx.x * blockDim.x + threadIdx.x;
    if (i < n) out[i] = (__bf16)in[i];
}

// W [K x N] row-major (fp32)  ->  Wt [N x K] row-major (bf16)  (column-major B)
__global__ __launch_bounds__(256) void transpose_to_bf16_kernel(
    const float* __restrict__ W, __bf16* __restrict__ Wt, int K, int N)
{
    int i = blockIdx.x * blockDim.x + threadIdx.x;
    if (i < K * N) {
        int k = i / N;
        int n = i - k * N;
        Wt[(size_t)n * K + k] = (__bf16)W[i];
    }
}

// ---------------------------------------------------------------------------
// Generic bf16 WMMA GEMM:  C[M x N] = A[M x 256] @ B[256 x N] + bias
//   A   : bf16 row-major, K = 256
//   Bt  : bf16, B stored column-major (Bt[n*256 + k])
//   C   : fp32 row-major
// Workgroup: 128 threads = 4 waves (2x2), each wave a 32x32 tile -> 64x64 blk.
// Fragment layouts follow CDNA5 ISA 7.12.2 exactly:
//   A 16-bit 16x32: lanes 0-15 kbase=0, lanes 16-31 kbase=8;
//                   VGPR 0-3 = K(kbase+0..7), VGPR 4-7 = K(kbase+16..23)
//   B 16-bit 32x16: same K order as A (lanes 0-15 kbase=0, lanes 16-31 kbase=8; K(kbase+0..7), K(kbase+16..23))
//   C/D f32 16x16 : VGPR r -> M=r (lanes 0-15), M=r+8 (lanes 16-31), N=lane&15
// ---------------------------------------------------------------------------
__global__ __launch_bounds__(128) void gemm_bf16_wmma_kernel(
    const __bf16* __restrict__ A,
    const __bf16* __restrict__ Bt,
    const float*  __restrict__ bias,
    float* __restrict__ C,
    int N)
{
    const int lane = threadIdx.x & 31;
    const int wave = threadIdx.x >> 5;
    const int wm   = wave >> 1;          // 0..1
    const int wn   = wave & 1;           // 0..1
    const int rowBase = blockIdx.x * 64 + wm * 32;
    const int colBase = blockIdx.y * 64 + wn * 32;

    const int l15 = lane & 15;
    const int lhi = lane >> 4;           // 0 or 1
    const int kbaseA = lhi * 8;          // A-fragment per-lane K offset
    const int kbaseB = lhi * 8;          // B-fragment per-lane K offset

    v8f acc[2][2] = {};

    for (int k0 = 0; k0 < DM; k0 += 32) {
        v16bf afrag[2], bfrag[2];
#pragma unroll
        for (int mt = 0; mt < 2; ++mt) {
            const __bf16* ap = A + (size_t)(rowBase + mt * 16 + l15) * DM + k0 + kbaseA;
            v8bf lo = *(const v8bf*)(ap);        // K = kbase + 0..7
            v8bf hi = *(const v8bf*)(ap + 16);   // K = kbase + 16..23
            afrag[mt] = __builtin_shufflevector(lo, hi,
                0,1,2,3,4,5,6,7,8,9,10,11,12,13,14,15);
        }
#pragma unroll
        for (int nt = 0; nt < 2; ++nt) {
            const __bf16* bp = Bt + (size_t)(colBase + nt * 16 + l15) * DM + k0 + kbaseB;
            v8bf lo = *(const v8bf*)(bp);        // K = kbase + 0..7
            v8bf hi = *(const v8bf*)(bp + 16);   // K = kbase + 16..23
            bfrag[nt] = __builtin_shufflevector(lo, hi,
                0,1,2,3,4,5,6,7,8,9,10,11,12,13,14,15);
        }
#pragma unroll
        for (int mt = 0; mt < 2; ++mt)
#pragma unroll
            for (int nt = 0; nt < 2; ++nt)
                acc[mt][nt] = __builtin_amdgcn_wmma_f32_16x16x32_bf16(
                    /*neg_a=*/false, afrag[mt],
                    /*neg_b=*/false, bfrag[nt],
                    /*c_mod=*/(short)0, acc[mt][nt],
                    /*reuse_a=*/false, /*reuse_b=*/false);
    }

    // Store D (+bias). VGPR r: M = r + lhi*8, N = l15.
#pragma unroll
    for (int mt = 0; mt < 2; ++mt) {
#pragma unroll
        for (int nt = 0; nt < 2; ++nt) {
            const int col = colBase + nt * 16 + l15;
            const float b = bias[col];
#pragma unroll
            for (int r = 0; r < 8; ++r) {
                const int row = rowBase + mt * 16 + lhi * 8 + r;
                C[(size_t)row * N + col] = acc[mt][nt][r] + b;
            }
        }
    }
}

// ---------------------------------------------------------------------------
// Softmax over 16 points per (query, head). In-place on attn [LQ*NHEADS, 16].
// ---------------------------------------------------------------------------
__global__ __launch_bounds__(256) void softmax16_kernel(float* __restrict__ attn, int total)
{
    int t = blockIdx.x * blockDim.x + threadIdx.x;
    if (t >= total) return;
    float* p = attn + (size_t)t * 16;
    float v[16];
    float mx = -3.402823e38f;
#pragma unroll
    for (int i = 0; i < 16; ++i) { v[i] = p[i]; mx = fmaxf(mx, v[i]); }
    float s = 0.f;
#pragma unroll
    for (int i = 0; i < 16; ++i) { v[i] = __expf(v[i] - mx); s += v[i]; }
    float inv = 1.f / s;
#pragma unroll
    for (int i = 0; i < 16; ++i) p[i] = v[i] * inv;
}

// ---------------------------------------------------------------------------
// Bilinear sampling + attention-weighted accumulation.
// One wave32 per (query, head); lane = channel (Dh = 32).
// value [LIN, 256] fp32 (channel = head*32 + d)  -- 22 MB, L2-resident.
// off   [LQ, 256]  fp32 : (((m*4+lvl)*4+p)*2+c)
// attn  [LQ, 128]  fp32 : (m*16 + lvl*4 + p), post-softmax
// ref   [LQ, 4, 2] fp32
// samp  [LQ, 256]  bf16 out (channel = head*32 + d)
// ---------------------------------------------------------------------------
__global__ __launch_bounds__(256) void msda_sample_kernel(
    const float* __restrict__ value,
    const float* __restrict__ off,
    const float* __restrict__ attn,
    const float* __restrict__ ref,
    __bf16* __restrict__ samp)
{
    const int lane = threadIdx.x & 31;
    const int wid  = blockIdx.x * (blockDim.x >> 5) + (threadIdx.x >> 5);
    const int q = wid >> 3;
    const int m = wid & 7;
    if (q >= LQ) return;

    const int   Hs[4]     = {128, 64, 32, 16};
    const int   Ws[4]     = {128, 64, 32, 16};
    const int   starts[4] = {0, 16384, 20480, 21504};

    const float* offq = off  + (size_t)q * 256 + m * 32;   // [4][4][2]
    const float* attq = attn + (size_t)q * 128 + m * 16;   // [16]
    const float* refq = ref  + (size_t)q * 8;              // [4][2]
    const int chan = m * DH + lane;

    float acc = 0.f;

#pragma unroll
    for (int lvl = 0; lvl < 4; ++lvl) {
        const int   H = Hs[lvl], W = Ws[lvl], st = starts[lvl];
        const float rx = refq[lvl * 2 + 0];
        const float ry = refq[lvl * 2 + 1];
        const float invW = 1.f / (float)W;
        const float invH = 1.f / (float)H;
#pragma unroll
        for (int p = 0; p < 4; ++p) {
            const float ox = offq[(lvl * 4 + p) * 2 + 0];
            const float oy = offq[(lvl * 4 + p) * 2 + 1];
            const float a  = attq[lvl * 4 + p];

            const float xn = rx + ox * invW;
            const float yn = ry + oy * invH;
            const float x  = xn * (float)W - 0.5f;
            const float y  = yn * (float)H - 0.5f;
            const float x0 = floorf(x), y0 = floorf(y);
            const float x1 = x0 + 1.f,  y1 = y0 + 1.f;

            const float wa = (x1 - x) * (y1 - y);
            const float wb = (x1 - x) * (y - y0);
            const float wc = (x - x0) * (y1 - y);
            const float wd = (x - x0) * (y - y0);

            const float x0v = (x0 >= 0.f && x0 <= (float)(W - 1)) ? 1.f : 0.f;
            const float x1v = (x1 >= 0.f && x1 <= (float)(W - 1)) ? 1.f : 0.f;
            const float y0v = (y0 >= 0.f && y0 <= (float)(H - 1)) ? 1.f : 0.f;
            const float y1v = (y1 >= 0.f && y1 <= (float)(H - 1)) ? 1.f : 0.f;

            int x0i = min(max((int)x0, 0), W - 1);
            int x1i = min(max((int)x1, 0), W - 1);
            int y0i = min(max((int)y0, 0), H - 1);
            int y1i = min(max((int)y1, 0), H - 1);

            // coalesced 128B gathers across the wave (lane = channel)
            const float va = value[(size_t)(st + y0i * W + x0i) * DM + chan];
            const float vb = value[(size_t)(st + y1i * W + x0i) * DM + chan];
            const float vc = value[(size_t)(st + y0i * W + x1i) * DM + chan];
            const float vd = value[(size_t)(st + y1i * W + x1i) * DM + chan];

            const float center =
                (xn >= 0.f && xn <= 1.f && yn >= 0.f && yn <= 1.f) ? 1.f : 0.f;

            float s = wa * va * (x0v * y0v)
                    + wb * vb * (x0v * y1v)
                    + wc * vc * (x1v * y0v)
                    + wd * vd * (x1v * y1v);
            acc += a * center * s;
        }
    }

    samp[(size_t)q * 256 + chan] = (__bf16)acc;
}

// ---------------------------------------------------------------------------
// Launcher
// ---------------------------------------------------------------------------
extern "C" void kernel_launch(void* const* d_in, const int* in_sizes, int n_in,
                              void* d_out, int out_size, void* d_ws, size_t ws_size,
                              hipStream_t stream) {
    (void)in_sizes; (void)n_in; (void)out_size; (void)ws_size;

    const float* query    = (const float*)d_in[0];   // [LQ, 256]
    const float* refpts   = (const float*)d_in[1];   // [LQ, 4, 2]
    const float* in_flat  = (const float*)d_in[2];   // [LIN, 256]
    /* d_in[3] = input_spatial_shapes (hardcoded) */
    const float* Woff     = (const float*)d_in[4];   // [256, 256]
    const float* boff     = (const float*)d_in[5];   // [256]
    const float* Wattn    = (const float*)d_in[6];   // [256, 128]
    const float* battn    = (const float*)d_in[7];   // [128]
    const float* Wval     = (const float*)d_in[8];   // [256, 256]
    const float* bval     = (const float*)d_in[9];   // [256]
    const float* Wout     = (const float*)d_in[10];  // [256, 256]
    const float* bout     = (const float*)d_in[11];  // [256]
    float* out            = (float*)d_out;           // [LQ, 256]

    // ---- workspace carve-out (256B aligned blocks) ----
    char* base = (char*)d_ws;
    size_t off_b = 0;
    auto carve = [&](size_t bytes) -> char* {
        char* p = base + off_b;
        off_b += (bytes + 255) & ~(size_t)255;
        return p;
    };
    __bf16* q_bf    = (__bf16*)carve((size_t)LQ  * DM  * 2);
    __bf16* in_bf   = (__bf16*)carve((size_t)LIN * DM  * 2);
    __bf16* wv_t    = (__bf16*)carve((size_t)DM  * DM  * 2);
    __bf16* woff_t  = (__bf16*)carve((size_t)DM  * DM  * 2);
    __bf16* wattn_t = (__bf16*)carve((size_t)DM  * 128 * 2);
    __bf16* wout_t  = (__bf16*)carve((size_t)DM  * DM  * 2);
    float*  value_f = (float*) carve((size_t)LIN * DM  * 4);
    float*  off_f   = (float*) carve((size_t)LQ  * DM  * 4);
    float*  attn_f  = (float*) carve((size_t)LQ  * 128 * 4);
    __bf16* samp_bf = (__bf16*)carve((size_t)LQ  * DM  * 2);

    const int NE = LQ * DM;                       // 5,570,560

    // 1) precision staging: fp32 -> bf16 (activations + transposed weights)
    f32_to_bf16_kernel<<<(NE + 255) / 256, 256, 0, stream>>>(query,   q_bf,  NE);
    f32_to_bf16_kernel<<<(NE + 255) / 256, 256, 0, stream>>>(in_flat, in_bf, NE);
    transpose_to_bf16_kernel<<<(DM * DM  + 255) / 256, 256, 0, stream>>>(Wval,  wv_t,    DM, DM);
    transpose_to_bf16_kernel<<<(DM * DM  + 255) / 256, 256, 0, stream>>>(Woff,  woff_t,  DM, DM);
    transpose_to_bf16_kernel<<<(DM * 128 + 255) / 256, 256, 0, stream>>>(Wattn, wattn_t, DM, 128);
    transpose_to_bf16_kernel<<<(DM * DM  + 255) / 256, 256, 0, stream>>>(Wout,  wout_t,  DM, DM);

    // 2) WMMA GEMMs: value proj, sampling offsets, attention logits
    dim3 blk(128);
    gemm_bf16_wmma_kernel<<<dim3(LQ / 64, 4), blk, 0, stream>>>(in_bf, wv_t,   bval,  value_f, DM);
    gemm_bf16_wmma_kernel<<<dim3(LQ / 64, 4), blk, 0, stream>>>(q_bf,  woff_t, boff,  off_f,   DM);
    gemm_bf16_wmma_kernel<<<dim3(LQ / 64, 2), blk, 0, stream>>>(q_bf,  wattn_t,battn, attn_f,  128);

    // 3) softmax over 16 (level,point) slots per (q, head)
    const int TOT_QH = LQ * NHEADS;               // 174,080
    softmax16_kernel<<<(TOT_QH + 255) / 256, 256, 0, stream>>>(attn_f, TOT_QH);

    // 4) bilinear sampling + weighted accumulation (wave per (q, head))
    msda_sample_kernel<<<LQ, 256, 0, stream>>>(value_f, off_f, attn_f, refpts, samp_bf);

    // 5) output projection -> d_out (fp32)
    gemm_bf16_wmma_kernel<<<dim3(LQ / 64, 4), blk, 0, stream>>>(samp_bf, wout_t, bout, out, DM);
}
